// DeepAveragingLSTMNetwork_11897059410611
// MI455X (gfx1250) — matrix operand order repaired and checked by reading, compiled, not run on hardware
//
#include <hip/hip_runtime.h>
#include <hip/hip_bf16.h>

// ---------------------------------------------------------------------------
// DeepAveragingLSTMNetwork for MI455X (gfx1250, wave32, WMMA bf16 16x16x32)
// ---------------------------------------------------------------------------

typedef __attribute__((ext_vector_type(16))) __bf16 v16bf;
typedef __attribute__((ext_vector_type(8)))  __bf16 v8bf;
typedef __attribute__((ext_vector_type(8)))  float  v8f;

#define N_WORDS 4096
#define LMAX    24
#define DW      300
#define DC      128
#define HDIM    512
#define G4      2048          // 4*H
#define KTOT    640           // DC + H
#define HID     512
#define OUTK    5

// workspace layout (bytes)
#define WS_WC_OFF    0                         // bf16 bits [2048][640]
#define WS_BIAS_OFF  (G4 * KTOT * 2)           // float [2048]
#define WS_SUM_OFF   (WS_BIAS_OFF + G4 * 4)    // float [1024] (812 used)

__device__ __forceinline__ unsigned short f2bf_bits(float f) {
  union { float f; unsigned u; } v; v.f = f;
  unsigned r = v.u + 0x7FFFu + ((v.u >> 16) & 1u);   // round-to-nearest-even
  return (unsigned short)(r >> 16);
}
__device__ __forceinline__ float bits2f(unsigned short s) {
  union { unsigned u; float f; } v; v.u = ((unsigned)s) << 16;
  return v.f;
}
__device__ __forceinline__ float fsigm(float x) { return 1.f / (1.f + __expf(-x)); }
__device__ __forceinline__ float ftanh(float x) { return 2.f / (1.f + __expf(-2.f * x)) - 1.f; }

union AFrag { v16bf v; v8bf h[2]; };

// ---------------------------------------------------------------------------
// Kernel 1: convert weights to bf16 combined [G4][KTOT]; bias = b_ih + b_hh;
//           zero the reduction accumulator.
// ---------------------------------------------------------------------------
__global__ void prep_kernel(const float* __restrict__ W_ih,
                            const float* __restrict__ W_hh,
                            const float* __restrict__ b_ih,
                            const float* __restrict__ b_hh,
                            unsigned short* __restrict__ Wc,
                            float* __restrict__ biasf,
                            float* __restrict__ sumv) {
  int idx = blockIdx.x * blockDim.x + threadIdx.x;
  if (idx < G4 * KTOT) {
    int g = idx / KTOT, k = idx % KTOT;
    float w = (k < DC) ? W_ih[g * DC + k] : W_hh[g * HDIM + (k - DC)];
    Wc[idx] = f2bf_bits(w);
  }
  if (idx < G4)  biasf[idx] = b_ih[idx] + b_hh[idx];
  if (idx < 1024) sumv[idx] = 0.f;
}

// ---------------------------------------------------------------------------
// Kernel 2: glove column sums (block j owns column j; no atomics needed).
// ---------------------------------------------------------------------------
__global__ void glove_sum_kernel(const int* __restrict__ widx,
                                 const float* __restrict__ glove,
                                 float* __restrict__ sumv) {
  __shared__ float red[256];
  const int j = blockIdx.x;          // 0..DW-1
  float s = 0.f;
  for (int i = threadIdx.x; i < N_WORDS; i += 256)
    s += glove[(size_t)widx[i] * DW + j];
  red[threadIdx.x] = s;
  __syncthreads();
  for (int off = 128; off > 0; off >>= 1) {
    if (threadIdx.x < off) red[threadIdx.x] += red[threadIdx.x + off];
    __syncthreads();
  }
  if (threadIdx.x == 0) sumv[j] = red[0];
}

// ---------------------------------------------------------------------------
// Kernel 3: LSTM over 24 steps. One WG = 16 words, 8 waves, each wave owns
// 64 hidden units (4 WMMA col-groups x 4 gates). h double-buffered in LDS
// (bf16), c resident in WMMA C-layout VGPRs. One barrier per step.
// K-loops are deliberately NOT unrolled to keep <=1 A-frag + 4 B-frags in
// flight (avoids VGPR spills / s_set_vgpr_msb seen with full unroll).
// ---------------------------------------------------------------------------
#define CE_PAD (DC + 8)        // 136
#define HB_PAD (HDIM + 8)      // 520

// compile-time byte-expressible element offsets for gate columns
#define GATE_STRIDE ((size_t)HDIM * KTOT)      // elements between gate blocks

__global__ __launch_bounds__(256) void lstm_kernel(
    const int* __restrict__ char_indices,      // [N, L]
    const int* __restrict__ char_lengths,      // [N]
    const float* __restrict__ char_embed,      // [100, DC] f32
    const unsigned short* __restrict__ Wc,     // [G4][KTOT] bf16 bits
    const float* __restrict__ biasf,           // [G4]
    float* __restrict__ sumv)                  // [812] accumulator
{
  __shared__ __align__(16) unsigned short cebuf[100][CE_PAD];     // char table
  __shared__ __align__(16) unsigned short hbuf[2][16][HB_PAD];    // h ping-pong
  __shared__ int lenBuf[16];
  __shared__ int cIdx[16][LMAX];

  const int tid     = threadIdx.x;
  const int wave    = tid >> 5;
  const int lane    = tid & 31;
  const int laneLo  = lane & 15;
  const int laneHi  = lane >> 4;
  const int rowBase = blockIdx.x * 16;

  // stage char-embedding table into LDS as bf16
  for (int i = tid; i < 100 * DC; i += 256) {
    int r = i / DC, c = i % DC;
    cebuf[r][c] = f2bf_bits(char_embed[i]);
  }
  // zero h buffers
  for (int i = tid; i < 16 * HDIM; i += 256) {
    int r = i / HDIM, c = i % HDIM;
    hbuf[0][r][c] = 0;
    hbuf[1][r][c] = 0;
  }
  if (tid < 16) lenBuf[tid] = char_lengths[rowBase + tid];
  for (int i = tid; i < 16 * LMAX; i += 256) {
    int r = i / LMAX, t = i % LMAX;
    cIdx[r][t] = char_indices[(rowBase + r) * LMAX + t];
  }
  __syncthreads();

  // preload bias for this wave's 64 units (per-lane: column N = laneLo)
  float biasr[4][4];
#pragma unroll
  for (int gi = 0; gi < 4; ++gi)
#pragma unroll
    for (int ga = 0; ga < 4; ++ga)
      biasr[gi][ga] = biasf[ga * HDIM + wave * 64 + gi * 16 + laneLo];

  // per-lane weight base: row = wave*64 + laneLo, K-phase = laneHi*16
  const unsigned short* const wbase =
      Wc + (size_t)(wave * 64 + laneLo) * KTOT + laneHi * 16;

  // cell state: 4 groups x WMMA C-tile (lane laneLo = col, reg r -> row)
  v8f cc[4];
#pragma unroll
  for (int gi = 0; gi < 4; ++gi)
#pragma unroll
    for (int k = 0; k < 8; ++k) cc[gi][k] = 0.f;

  for (int t = 0; t < LMAX; ++t) {
    const int cur = t & 1, nxt = cur ^ 1;
    const int myCe = cIdx[laneLo][t];   // char row for A-matrix row M=laneLo

#pragma unroll
    for (int gi = 0; gi < 4; ++gi) {
      const int ub = wave * 64 + gi * 16;

      v8f acc0, acc1, acc2, acc3;
#pragma unroll
      for (int k = 0; k < 8; ++k) {
        acc0[k] = biasr[gi][0];
        acc1[k] = biasr[gi][1];
        acc2[k] = biasr[gi][2];
        acc3[k] = biasr[gi][3];
      }

      // single per-lane pointer; gate blocks reached via constant offsets
      const unsigned short* bp = wbase + (size_t)gi * 16 * KTOT;
      const unsigned short* ap = &cebuf[myCe][laneHi * 16];
      const unsigned short* hp = &hbuf[cur][laneLo][laneHi * 16];

      // ---- x-part: K tiles 0..3 from char-embedding table -----------------
#pragma unroll 1
      for (int kt = 0; kt < 4; ++kt) {
        AFrag a;
        a.h[0] = *(const v8bf*)(ap);
        a.h[1] = *(const v8bf*)(ap + 8);
        ap += 32;
        v16bf b0 = *(const v16bf*)(bp);
        v16bf b1 = *(const v16bf*)(bp + GATE_STRIDE);
        v16bf b2 = *(const v16bf*)(bp + 2 * GATE_STRIDE);
        v16bf b3 = *(const v16bf*)(bp + 3 * GATE_STRIDE);
        bp += 32;
        acc0 = __builtin_amdgcn_wmma_f32_16x16x32_bf16(false, a.v, false, b0, (short)0, acc0, false, false);
        acc1 = __builtin_amdgcn_wmma_f32_16x16x32_bf16(false, a.v, false, b1, (short)0, acc1, false, false);
        acc2 = __builtin_amdgcn_wmma_f32_16x16x32_bf16(false, a.v, false, b2, (short)0, acc2, false, false);
        acc3 = __builtin_amdgcn_wmma_f32_16x16x32_bf16(false, a.v, false, b3, (short)0, acc3, false, false);
      }

      // ---- h-part: K tiles 4..19 from recurrent state ---------------------
#pragma unroll 1
      for (int kt = 0; kt < 16; ++kt) {
        AFrag a;
        a.h[0] = *(const v8bf*)(hp);
        a.h[1] = *(const v8bf*)(hp + 8);
        hp += 32;
        v16bf b0 = *(const v16bf*)(bp);
        v16bf b1 = *(const v16bf*)(bp + GATE_STRIDE);
        v16bf b2 = *(const v16bf*)(bp + 2 * GATE_STRIDE);
        v16bf b3 = *(const v16bf*)(bp + 3 * GATE_STRIDE);
        bp += 32;
        acc0 = __builtin_amdgcn_wmma_f32_16x16x32_bf16(false, a.v, false, b0, (short)0, acc0, false, false);
        acc1 = __builtin_amdgcn_wmma_f32_16x16x32_bf16(false, a.v, false, b1, (short)0, acc1, false, false);
        acc2 = __builtin_amdgcn_wmma_f32_16x16x32_bf16(false, a.v, false, b2, (short)0, acc2, false, false);
        acc3 = __builtin_amdgcn_wmma_f32_16x16x32_bf16(false, a.v, false, b3, (short)0, acc3, false, false);
      }

      // LSTM cell elementwise for this 16-unit group (C-layout coords)
#pragma unroll
      for (int r = 0; r < 8; ++r) {
        const int M = r + laneHi * 8;
        float ig = fsigm(acc0[r]);
        float fg = fsigm(acc1[r]);
        float gg = ftanh(acc2[r]);
        float og = fsigm(acc3[r]);
        float cn = fg * cc[gi][r] + ig * gg;
        float hn = og * ftanh(cn);
        bool valid = (t < lenBuf[M]);
        float hold = bits2f(hbuf[cur][M][ub + laneLo]);
        cc[gi][r] = valid ? cn : cc[gi][r];
        hbuf[nxt][M][ub + laneLo] = f2bf_bits(valid ? hn : hold);
      }
    }
    __syncthreads();
  }

  // h_last is in hbuf[0] (t=23: nxt = 0). Mask len>=2 and reduce over rows.
  for (int u = tid; u < HDIM; u += 256) {
    float s = 0.f;
#pragma unroll
    for (int r = 0; r < 16; ++r)
      if (lenBuf[r] >= 2) s += bits2f(hbuf[0][r][u]);
    unsafeAtomicAdd(&sumv[DW + u], s);
  }
}

// ---------------------------------------------------------------------------
// Kernel 4: head: avg -> fc1 -> sigmoid -> fc2 -> logits[5]
// ---------------------------------------------------------------------------
__global__ void head_kernel(const float* __restrict__ sumv,
                            const float* __restrict__ fc1W,
                            const float* __restrict__ fc1b,
                            const float* __restrict__ fc2W,
                            const float* __restrict__ fc2b,
                            float* __restrict__ out) {
  __shared__ float h1[HID];
  const float invN = 1.0f / (float)N_WORDS;
  const int i = threadIdx.x;    // 512 threads
  if (i < HID) {
    float s = fc1b[i];
    const float* w = fc1W + (size_t)i * (DW + HDIM);
    for (int k = 0; k < DW + HDIM; ++k) s += (sumv[k] * invN) * w[k];
    h1[i] = fsigm(s);
  }
  __syncthreads();
  if (i < OUTK) {
    float s = fc2b[i];
    const float* w = fc2W + (size_t)i * HID;
    for (int k = 0; k < HID; ++k) s += h1[k] * w[k];
    out[i] = s;
  }
}

// ---------------------------------------------------------------------------
extern "C" void kernel_launch(void* const* d_in, const int* in_sizes, int n_in,
                              void* d_out, int out_size, void* d_ws, size_t ws_size,
                              hipStream_t stream) {
  (void)in_sizes; (void)n_in; (void)out_size; (void)ws_size;

  const int*   word_indices = (const int*)d_in[0];
  const int*   char_indices = (const int*)d_in[1];
  const int*   char_lengths = (const int*)d_in[2];
  const float* glove_table  = (const float*)d_in[3];
  const float* char_embed   = (const float*)d_in[4];
  const float* W_ih  = (const float*)d_in[5];
  const float* W_hh  = (const float*)d_in[6];
  const float* b_ih  = (const float*)d_in[7];
  const float* b_hh  = (const float*)d_in[8];
  const float* fc1_W = (const float*)d_in[9];
  const float* fc1_b = (const float*)d_in[10];
  const float* fc2_W = (const float*)d_in[11];
  const float* fc2_b = (const float*)d_in[12];
  float* out = (float*)d_out;

  char* ws = (char*)d_ws;
  unsigned short* Wc    = (unsigned short*)(ws + WS_WC_OFF);
  float*          biasf = (float*)(ws + WS_BIAS_OFF);
  float*          sumv  = (float*)(ws + WS_SUM_OFF);

  prep_kernel<<<(G4 * KTOT + 255) / 256, 256, 0, stream>>>(
      W_ih, W_hh, b_ih, b_hh, Wc, biasf, sumv);
  glove_sum_kernel<<<DW, 256, 0, stream>>>(word_indices, glove_table, sumv);
  lstm_kernel<<<N_WORDS / 16, 256, 0, stream>>>(
      char_indices, char_lengths, char_embed, Wc, biasf, sumv);
  head_kernel<<<1, 512, 0, stream>>>(sumv, fc1_W, fc1_b, fc2_W, fc2_b, out);
}
